// KnowledgeRNN_54288386621721
// MI455X (gfx1250) — compile-verified, matrix operand order checked by reading, not run on hardware
//
#include <hip/hip_runtime.h>
#include <hip/hip_bf16.h>
#include <math.h>

typedef __attribute__((ext_vector_type(16))) __bf16 bf16x16;
typedef __attribute__((ext_vector_type(8)))  float  f32x8;
typedef unsigned int u32x4 __attribute__((ext_vector_type(4)));
typedef int          i32x4 __attribute__((ext_vector_type(4)));
typedef int          i32x8 __attribute__((ext_vector_type(8)));

#define T_SEQ 2048
#define HID   1024
#define EMB   512
#define NG    4096        // 4*H gate rows
#define NTOK  33278
#define NTOK_PAD 33280    // padded to multiple of 16

// ---------- helpers ----------
__device__ __forceinline__ unsigned short f2bf(float f) {
  unsigned u = __float_as_uint(f);
  u += 0x7fffu + ((u >> 16) & 1u);          // round-to-nearest-even
  return (unsigned short)(u >> 16);
}
__device__ __forceinline__ float bf2f(unsigned u) {   // u must be <= 0xffff
  return __uint_as_float(u << 16);
}

union FragU {            // WMMA 16-bit A/B fragment: 16 bf16 = 2 x b128 loads
  bf16x16 v;
  uint4   q[2];
};

// ---------- Tensor Data Mover: 2D bf16 tile  Global -> LDS ----------
// D# built per CDNA5 ISA 8.3/8.4. tile = tile_d1 rows x tile_d0 elems,
// row-major tensor with row stride stride0 (elements), data_size = 2 bytes.
__device__ __forceinline__ void tdm_load_2d_bf16(unsigned lds_off,
                                                 const void* gptr,
                                                 unsigned tensor_d0,
                                                 unsigned tensor_d1,
                                                 unsigned tile_d0,
                                                 unsigned tile_d1,
                                                 unsigned stride0) {
  unsigned long long ga = (unsigned long long)(size_t)gptr;
  u32x4 g0;
  g0[0] = 0x1u;                                    // count=1 (valid), user mode
  g0[1] = lds_off;                                 // lds_addr (bytes)
  g0[2] = (unsigned)(ga & 0xffffffffu);            // global_addr[31:0]
  g0[3] = (unsigned)((ga >> 32) & 0x1ffffffu)      // global_addr[56:32]
        | (2u << 30);                              // type = 2 ("image")
  i32x8 g1;
  g1[0] = (int)(1u << 16);                         // data_size=1 -> 2 bytes
  g1[1] = (int)((tensor_d0 & 0xffffu) << 16);      // tensor_dim0[15:0]
  g1[2] = (int)((tensor_d0 >> 16) | ((tensor_d1 & 0xffffu) << 16));
  g1[3] = (int)((tensor_d1 >> 16) | (tile_d0 << 16));   // tile_dim0
  g1[4] = (int)(tile_d1 & 0xffffu);                // tile_dim1 (tile_dim2=0)
  g1[5] = (int)stride0;                            // tensor_dim0_stride[31:0]
  g1[6] = 0;                                       // stride hi / dim1_stride lo
  g1[7] = 0;
  i32x4 z4 = {0, 0, 0, 0};
#if defined(__clang_major__) && (__clang_major__ >= 23)
  i32x8 z8 = {0, 0, 0, 0, 0, 0, 0, 0};
  __builtin_amdgcn_tensor_load_to_lds(g0, g1, z4, z4, z8, 0);
#else
  __builtin_amdgcn_tensor_load_to_lds(g0, g1, z4, z4, 0);
#endif
}

// ---------- conversion kernels ----------
__global__ void k_f32_to_bf16(const float* __restrict__ src,
                              unsigned short* __restrict__ dst, int n) {
  int i = blockIdx.x * blockDim.x + threadIdx.x;
  int stride = gridDim.x * blockDim.x;
  for (; i < n; i += stride) dst[i] = f2bf(src[i]);
}

__global__ void k_gather_embed(const int* __restrict__ tokens,
                               const float* __restrict__ emb,
                               unsigned short* __restrict__ xb) {
  int t = blockIdx.x;
  int tok = tokens[t];
  const float* row = emb + (size_t)tok * EMB;
  for (int e = threadIdx.x; e < EMB; e += blockDim.x)
    xb[(size_t)t * EMB + e] = f2bf(row[e]);
}

// ---------- GEMM 1: x_gates = X(bf16) @ W_ih^T + b_ih + b_hh ----------
// Block: 8 waves, each a 16(M)x64(N) tile over shared N-range; K = 512.
// B tile (64 rows x 512 K = 64 KB) staged into LDS by the TDM once per block.
__global__ __launch_bounds__(256)
void k_gemm_xgates(const unsigned short* __restrict__ A,   // [2048,512] bf16
                   const unsigned short* __restrict__ B,   // [4096,512] bf16
                   const float* __restrict__ b_ih,
                   const float* __restrict__ b_hh,
                   float* __restrict__ C) {                // [2048,4096] f32
  __shared__ unsigned short Btile[64 * EMB];               // 64 KB
  const int lane = threadIdx.x & 31;
  const int wave = threadIdx.x >> 5;
  const int mt   = blockIdx.y * 8 + wave;
  const int n0   = blockIdx.x * 64;
  const int mrow = lane & 15, hi = lane >> 4;

  if (threadIdx.x < 32) {
    tdm_load_2d_bf16((unsigned)(size_t)Btile, B + (size_t)n0 * EMB,
                     EMB, NG, EMB, 64, EMB);
  }
  const unsigned short* arow = A + (size_t)(mt * 16 + mrow) * EMB;
  FragU a_cur, a_nxt;
  a_cur.q[0] = *(const uint4*)(arow + hi * 8);
  a_cur.q[1] = *(const uint4*)(arow + 16 + hi * 8);
  a_nxt = a_cur;
  if (threadIdx.x < 32) __builtin_amdgcn_s_wait_tensorcnt((short)0);
  __syncthreads();

  const f32x8 zero = {0.f,0.f,0.f,0.f,0.f,0.f,0.f,0.f};
  f32x8 acc[4] = {zero, zero, zero, zero};

  for (int kc = 0; kc < EMB / 32; ++kc) {
    if (kc + 1 < EMB / 32) {     // double-buffer A from global
      a_nxt.q[0] = *(const uint4*)(arow + (kc + 1) * 32 + hi * 8);
      a_nxt.q[1] = *(const uint4*)(arow + (kc + 1) * 32 + 16 + hi * 8);
    }
#pragma unroll
    for (int s = 0; s < 4; ++s) {
      const unsigned short* brow = &Btile[(s * 16 + mrow) * EMB];
      FragU b;
      b.q[0] = *(const uint4*)(brow + kc * 32 + hi * 8);
      b.q[1] = *(const uint4*)(brow + kc * 32 + 16 + hi * 8);
      acc[s] = __builtin_amdgcn_wmma_f32_16x16x32_bf16(
          false, a_cur.v, false, b.v, (short)0, acc[s], false, false);
    }
    a_cur = a_nxt;
  }
#pragma unroll
  for (int s = 0; s < 4; ++s) {
    int n = n0 + s * 16 + mrow;
    float bias = b_ih[n] + b_hh[n];
#pragma unroll
    for (int j = 0; j < 8; ++j)
      C[(size_t)(mt * 16 + hi * 8 + j) * NG + n] = acc[s][j] + bias;
  }
}

// ---------- LSTM persistent scan ----------
// 64 workgroups; WG b owns hidden indices [b*16, b*16+16).
// Gates for those indices = 64 dots of length 1024/step, 4 threads per dot.
// c kept in LDS, h broadcast via global (L2), W_hh bf16 is L2-resident.
__global__ __launch_bounds__(256)
void k_lstm_scan(const float* __restrict__ xg,             // [T,4096]
                 const unsigned short* __restrict__ Whh,   // [4096,1024] bf16
                 float* __restrict__ hglob,                // [1024] f32
                 unsigned short* __restrict__ hsb,         // [T,1024] bf16
                 unsigned* __restrict__ bar) {             // 2 counters, zeroed
  __shared__ float h_lds[HID];
  __shared__ float gate_lds[64];
  __shared__ float c_lds[16];

  const int tid = threadIdx.x;
  const int n0  = blockIdx.x * 16;
  const int r   = tid >> 2;                 // 0..63 : gate-row within slice
  const int q   = tid & 3;                  // K quarter
  const int g   = r >> 4, ii = r & 15;      // gate, local hidden idx
  const unsigned short* wrow =
      Whh + (size_t)(g * HID + n0 + ii) * HID + q * 256;

  for (int k = tid; k < HID; k += 256) h_lds[k] = 0.f;
  if (tid < 16) c_lds[tid] = 0.f;
  __syncthreads();

  unsigned gen = 0;
  for (int t = 0; t < T_SEQ; ++t) {
    float sum = 0.f;
#pragma unroll 4
    for (int kk = 0; kk < 256; kk += 8) {
      uint4 w = *(const uint4*)(wrow + kk);
      const float* hp = &h_lds[q * 256 + kk];
      sum += bf2f(w.x & 0xffffu) * hp[0];
      sum += bf2f(w.x >> 16)     * hp[1];
      sum += bf2f(w.y & 0xffffu) * hp[2];
      sum += bf2f(w.y >> 16)     * hp[3];
      sum += bf2f(w.z & 0xffffu) * hp[4];
      sum += bf2f(w.z >> 16)     * hp[5];
      sum += bf2f(w.w & 0xffffu) * hp[6];
      sum += bf2f(w.w >> 16)     * hp[7];
    }
    sum += __shfl_xor(sum, 1);
    sum += __shfl_xor(sum, 2);
    if (q == 0) gate_lds[r] = sum;
    __syncthreads();

    if (tid < 16) {
      const float* xgt = xg + (size_t)t * NG;
      int n = n0 + tid;
      float gi = xgt[n]            + gate_lds[tid];
      float gf = xgt[HID + n]      + gate_lds[16 + tid];
      float gg = xgt[2 * HID + n]  + gate_lds[32 + tid];
      float go = xgt[3 * HID + n]  + gate_lds[48 + tid];
      gi = 1.f / (1.f + __expf(-gi));
      gf = 1.f / (1.f + __expf(-gf));
      gg = tanhf(gg);
      go = 1.f / (1.f + __expf(-go));
      float c = gf * c_lds[tid] + gi * gg;
      c_lds[tid] = c;
      float h = go * tanhf(c);
      hglob[n] = h;
      hsb[(size_t)t * HID + n] = f2bf(h);
    }
    __syncthreads();

    // sense-reversing grid barrier (64 co-resident blocks)
    if (tid == 0) {
      __threadfence();
      unsigned arrived = atomicAdd(&bar[0], 1u);
      if (arrived == gridDim.x - 1) {
        __hip_atomic_store(&bar[0], 0u, __ATOMIC_RELAXED, __HIP_MEMORY_SCOPE_AGENT);
        __threadfence();
        atomicAdd(&bar[1], 1u);
      } else {
        while (__hip_atomic_load(&bar[1], __ATOMIC_ACQUIRE,
                                 __HIP_MEMORY_SCOPE_AGENT) == gen) {
          __builtin_amdgcn_s_sleep(2);
        }
      }
      gen++;
    }
    __syncthreads();
    for (int k = tid; k < HID; k += 256) h_lds[k] = hglob[k];
    __syncthreads();
  }
}

// ---------- GEMM 2: logits = hs(bf16) @ W_dec^T + b_dec  (into d_out) ----------
// Block: 8 waves, each 16(M)x32(N); K = 1024.
// B tile (32 rows x 1024 K = 64 KB) staged into LDS by the TDM.
__global__ __launch_bounds__(256)
void k_gemm_decoder(const unsigned short* __restrict__ A,  // [2048,1024] bf16
                    const unsigned short* __restrict__ B,  // [33280,1024] bf16
                    const float* __restrict__ b_dec,
                    float* __restrict__ out) {             // [2048,33278] f32
  __shared__ unsigned short Btile[32 * HID];               // 64 KB
  const int lane = threadIdx.x & 31;
  const int wave = threadIdx.x >> 5;
  const int mt   = blockIdx.y * 8 + wave;
  const int n0   = blockIdx.x * 32;
  const int mrow = lane & 15, hi = lane >> 4;

  if (threadIdx.x < 32) {
    tdm_load_2d_bf16((unsigned)(size_t)Btile, B + (size_t)n0 * HID,
                     HID, NTOK_PAD, HID, 32, HID);
  }
  const unsigned short* arow = A + (size_t)(mt * 16 + mrow) * HID;
  FragU a_cur, a_nxt;
  a_cur.q[0] = *(const uint4*)(arow + hi * 8);
  a_cur.q[1] = *(const uint4*)(arow + 16 + hi * 8);
  a_nxt = a_cur;
  if (threadIdx.x < 32) __builtin_amdgcn_s_wait_tensorcnt((short)0);
  __syncthreads();

  const f32x8 zero = {0.f,0.f,0.f,0.f,0.f,0.f,0.f,0.f};
  f32x8 acc[2] = {zero, zero};

  for (int kc = 0; kc < HID / 32; ++kc) {
    if (kc + 1 < HID / 32) {     // double-buffer A from global
      a_nxt.q[0] = *(const uint4*)(arow + (kc + 1) * 32 + hi * 8);
      a_nxt.q[1] = *(const uint4*)(arow + (kc + 1) * 32 + 16 + hi * 8);
    }
#pragma unroll
    for (int s = 0; s < 2; ++s) {
      const unsigned short* brow = &Btile[(s * 16 + mrow) * HID];
      FragU b;
      b.q[0] = *(const uint4*)(brow + kc * 32 + hi * 8);
      b.q[1] = *(const uint4*)(brow + kc * 32 + 16 + hi * 8);
      acc[s] = __builtin_amdgcn_wmma_f32_16x16x32_bf16(
          false, a_cur.v, false, b.v, (short)0, acc[s], false, false);
    }
    a_cur = a_nxt;
  }
#pragma unroll
  for (int s = 0; s < 2; ++s) {
    int n = n0 + s * 16 + mrow;
    if (n < NTOK) {
      float bd = b_dec[n];
#pragma unroll
      for (int j = 0; j < 8; ++j)
        out[(size_t)(mt * 16 + hi * 8 + j) * NTOK + n] = acc[s][j] + bd;
    }
  }
}

// ---------- fused in-place log_softmax over each row of d_out ----------
__global__ __launch_bounds__(256)
void k_logsoftmax(float* __restrict__ out) {
  __shared__ float red[8];
  const int t = blockIdx.x;
  float* row = out + (size_t)t * NTOK;
  const int tid = threadIdx.x, lane = tid & 31, wv = tid >> 5;

  float m = -INFINITY;
  for (int v = tid; v < NTOK; v += 256) m = fmaxf(m, row[v]);
#pragma unroll
  for (int o = 16; o; o >>= 1) m = fmaxf(m, __shfl_xor(m, o));
  if (lane == 0) red[wv] = m;
  __syncthreads();
  if (tid == 0) {
    float mm = red[0];
    for (int i = 1; i < 8; ++i) mm = fmaxf(mm, red[i]);
    red[0] = mm;
  }
  __syncthreads();
  m = red[0];

  float s = 0.f;
  for (int v = tid; v < NTOK; v += 256) s += __expf(row[v] - m);
#pragma unroll
  for (int o = 16; o; o >>= 1) s += __shfl_xor(s, o);
  __syncthreads();
  if (lane == 0) red[wv] = s;
  __syncthreads();
  if (tid == 0) {
    float ss = 0.f;
    for (int i = 0; i < 8; ++i) ss += red[i];
    red[0] = m + __logf(ss);
  }
  __syncthreads();
  float lse = red[0];
  for (int v = tid; v < NTOK; v += 256) row[v] -= lse;
}

// ---------- host ----------
extern "C" void kernel_launch(void* const* d_in, const int* in_sizes, int n_in,
                              void* d_out, int out_size, void* d_ws, size_t ws_size,
                              hipStream_t stream) {
  (void)in_sizes; (void)n_in; (void)out_size; (void)ws_size;
  const int*   tokens = (const int*)d_in[0];
  const float* emb    = (const float*)d_in[1];
  const float* W_ih   = (const float*)d_in[2];
  const float* W_hh   = (const float*)d_in[3];
  const float* b_ih   = (const float*)d_in[4];
  const float* b_hh   = (const float*)d_in[5];
  const float* W_dec  = (const float*)d_in[6];
  const float* b_dec  = (const float*)d_in[7];
  float* out = (float*)d_out;

  char* ws = (char*)d_ws;
  size_t off = 0;
  auto take = [&](size_t bytes) {
    char* p = ws + off;
    off = (off + bytes + 255) & ~(size_t)255;
    return p;
  };
  unsigned short* xb    = (unsigned short*)take((size_t)T_SEQ * EMB * 2);
  unsigned short* wihb  = (unsigned short*)take((size_t)NG * EMB * 2);
  unsigned short* whhb  = (unsigned short*)take((size_t)NG * HID * 2);
  unsigned short* wdecb = (unsigned short*)take((size_t)NTOK_PAD * HID * 2);
  float*          xg    = (float*)take((size_t)T_SEQ * NG * 4);
  float*          hbuf  = (float*)take((size_t)HID * 4);
  unsigned short* hsb   = (unsigned short*)take((size_t)T_SEQ * HID * 2);
  unsigned*       bar   = (unsigned*)take(256);

  // one-time (per launch) fp32 -> bf16 conversions
  k_f32_to_bf16<<<512,  256, 0, stream>>>(W_ih,  wihb, NG * EMB);
  k_f32_to_bf16<<<1024, 256, 0, stream>>>(W_hh,  whhb, NG * HID);
  k_f32_to_bf16<<<4096, 256, 0, stream>>>(W_dec, wdecb, NTOK * HID);
  hipMemsetAsync(wdecb + (size_t)NTOK * HID, 0,
                 (size_t)(NTOK_PAD - NTOK) * HID * 2, stream);
  k_gather_embed<<<T_SEQ, 256, 0, stream>>>(tokens, emb, xb);
  hipMemsetAsync(bar, 0, 8, stream);

  // x_gates = X @ W_ih^T + b_ih + b_hh   (WMMA bf16, TDM-staged B)
  dim3 g1(NG / 64, (T_SEQ / 16) / 8);
  k_gemm_xgates<<<g1, 256, 0, stream>>>(xb, wihb, b_ih, b_hh, xg);

  // sequential LSTM scan (persistent, grid-synced)
  k_lstm_scan<<<64, 256, 0, stream>>>(xg, whhb, hbuf, hsb, bar);

  // logits = hs @ W_dec^T + b_dec   (WMMA bf16, TDM-staged B, writes d_out)
  dim3 g2(NTOK_PAD / 32, (T_SEQ / 16) / 8);
  k_gemm_decoder<<<g2, 256, 0, stream>>>(hsb, wdecb, b_dec, out);

  // in-place log_softmax per row
  k_logsoftmax<<<T_SEQ, 256, 0, stream>>>(out);
}
